// MultiHeadAttention_35476429865984
// MI455X (gfx1250) — matrix operand order repaired and checked by reading, not compiled, not run on hardware
//
#include <hip/hip_runtime.h>
#include <hip/hip_bf16.h>
#include <stdint.h>

#define DIMD 4096
#define NH   64

typedef __attribute__((ext_vector_type(16))) __bf16         v16bf;
typedef __attribute__((ext_vector_type(8)))  float          v8f;
typedef __attribute__((ext_vector_type(16))) unsigned short v16u;
typedef __attribute__((ext_vector_type(8)))  unsigned short v8u;
typedef __attribute__((ext_vector_type(4)))  float          v4f;

static __device__ __forceinline__ unsigned short f2bf(float f) {
  union { float f; unsigned u; } x; x.f = f;
  unsigned r = x.u + 0x7FFFu + ((x.u >> 16) & 1u);  // round-to-nearest-even
  return (unsigned short)(r >> 16);
}

static __device__ __forceinline__ v16bf as_bf16(v16u u) {
  union { v16u u; v16bf b; } t; t.u = u; return t.b;
}

// CDNA5 async global->LDS copy (16B per lane), tracked by ASYNCcnt.
static __device__ __forceinline__ void async_b128(const void* g, const void* l) {
  unsigned lds = (unsigned)(unsigned long long)l;  // low 32 bits of generic shared ptr = LDS offset
  asm volatile("global_load_async_to_lds_b128 %0, %1, off" :: "v"(lds), "v"(g) : "memory");
}
// Async loads complete in order: waiting ASYNCcnt<=N releases the oldest copies.
template <int N>
static __device__ __forceinline__ void wait_async() {
  asm volatile("s_wait_asynccnt %0" :: "n"(N) : "memory");
}

// A-fragment (16x32 bf16, MxK): lane L = row (L&15); 16 contiguous K values
// starting at kb = (L>>4)*16 within a 32-wide K step.  p points at [row][kb].
static __device__ __forceinline__ v16bf frag_a(const unsigned short* p) {
  v8u lo = *(const v8u*)p;
  v8u hi = *(const v8u*)(p + 8);
  return as_bf16(__builtin_shufflevector(lo, hi, 0,1,2,3,4,5,6,7,8,9,10,11,12,13,14,15));
}
// B-fragment (32x16 bf16, KxN): lane L = col (L&15); 16 contiguous K values
// starting at kb = (L>>4)*16.  p points at [col_row][kb] (K-contiguous storage).
static __device__ __forceinline__ v16bf frag_b(const unsigned short* p) {
  v8u lo = *(const v8u*)p;
  v8u hi = *(const v8u*)(p + 8);
  return as_bf16(__builtin_shufflevector(lo, hi, 0,1,2,3,4,5,6,7,8,9,10,11,12,13,14,15));
}

__global__ void __launch_bounds__(256) cvt_f32_bf16(const float* __restrict__ in,
                                                    unsigned short* __restrict__ out, int n) {
  int i = (blockIdx.x * 256 + threadIdx.x) * 8;
  if (i + 8 <= n) {
    v4f a = *(const v4f*)(in + i);
    v4f b = *(const v4f*)(in + i + 4);
    v8u o;
    o[0]=f2bf(a.x); o[1]=f2bf(a.y); o[2]=f2bf(a.z); o[3]=f2bf(a.w);
    o[4]=f2bf(b.x); o[5]=f2bf(b.y); o[6]=f2bf(b.z); o[7]=f2bf(b.w);
    *(v8u*)(out + i) = o;
  }
}

// C[M,N] = A[M,K] * W[N,K]^T + bias[N];  M=N=K=4096, bf16 in, f32 accumulate.
// Block: 256 thr (8 waves as 2x4), block tile 128x128, K-step 64, double-buffered
// async global->LDS staging.  Per-wave tile 64x32 -> 16 WMMA per K-step.
template <bool BF16OUT>
__global__ void __launch_bounds__(256) gemm_xwt(const unsigned short* __restrict__ A,
                                                const unsigned short* __restrict__ W,
                                                const float* __restrict__ bias,
                                                void* __restrict__ Cout) {
  __shared__ __align__(16) unsigned short At[2][128 * 64];   // 2 x 16KB
  __shared__ __align__(16) unsigned short Bt[2][128 * 64];   // 2 x 16KB
  const int tid  = threadIdx.x;
  const int lane = tid & 31;
  const int wave = tid >> 5;
  const int wx = wave >> 2;            // 0..1 along M (64 rows each)
  const int wy = wave & 3;             // 0..3 along N (32 cols each)
  const int m0 = blockIdx.x * 128;
  const int n0 = blockIdx.y * 128;

  v8f acc[4][2] = {};

  // Issue 8 async b128 copies (4 A-chunks + 4 B-chunks) for one 128x64 K-panel.
  auto stage = [&](int nb, int kk) {
#pragma unroll
    for (int i = 0; i < 4; ++i) {
      int c = i * 256 + tid;
      int row = c >> 3, co = c & 7;
      async_b128(A + (size_t)(m0 + row) * DIMD + kk + co * 8, &At[nb][row * 64 + co * 8]);
      async_b128(W + (size_t)(n0 + row) * DIMD + kk + co * 8, &Bt[nb][row * 64 + co * 8]);
    }
  };

  stage(0, 0);
  int cur = 0;
  for (int k0 = 0; k0 < DIMD; k0 += 64) {
    if (k0 + 64 < DIMD) {
      stage(cur ^ 1, k0 + 64);   // overlap next panel's DMA with this panel's math
      wait_async<8>();           // oldest 8 (current panel) have landed
    } else {
      wait_async<0>();
    }
    __syncthreads();

#pragma unroll
    for (int ks = 0; ks < 2; ++ks) {
      v16bf af[4], bf[2];
      const int kb8  = ks * 32 + (lane >> 4) * 8;
      const int kb16 = ks * 32 + (lane >> 4) * 16;
#pragma unroll
      for (int rm = 0; rm < 4; ++rm)
        af[rm] = frag_a(&At[cur][(wx * 64 + rm * 16 + (lane & 15)) * 64 + kb16]);
#pragma unroll
      for (int cn = 0; cn < 2; ++cn)
        bf[cn] = frag_b(&Bt[cur][(wy * 32 + cn * 16 + (lane & 15)) * 64 + kb16]);
#pragma unroll
      for (int rm = 0; rm < 4; ++rm)
#pragma unroll
        for (int cn = 0; cn < 2; ++cn)
          acc[rm][cn] = __builtin_amdgcn_wmma_f32_16x16x32_bf16(
              false, af[rm], false, bf[cn], (short)0, acc[rm][cn], false, false);
    }
    __syncthreads();             // reads done before this buffer is overwritten
    cur ^= 1;
  }

#pragma unroll
  for (int rm = 0; rm < 4; ++rm)
#pragma unroll
    for (int cn = 0; cn < 2; ++cn) {
      int col = n0 + wy * 32 + cn * 16 + (lane & 15);
      float bv = bias[col];
#pragma unroll
      for (int r = 0; r < 8; ++r) {
        int row = m0 + wx * 64 + rm * 16 + ((lane >> 4) << 3) + r;
        float v = acc[rm][cn][r] + bv;
        if (BF16OUT) ((unsigned short*)Cout)[(size_t)row * DIMD + col] = f2bf(v);
        else         ((float*)Cout)[(size_t)row * DIMD + col] = v;
      }
    }
}

// One block per (head hi_, batch bb): scores=Q Kt, softmax (no 1/sqrt scale),
// out = P*V, then scatter per the reference's buggy cat/view regrouping.
__global__ void __launch_bounds__(128) attn_kernel(const unsigned short* __restrict__ Q,
                                                   const unsigned short* __restrict__ K,
                                                   const unsigned short* __restrict__ V,
                                                   unsigned short* __restrict__ Y) {
  __shared__ __align__(16) unsigned short Qs[64 * 64];
  __shared__ __align__(16) unsigned short Ks[64 * 64];
  __shared__ __align__(16) unsigned short Ps[64 * 64];
  __shared__ __align__(16) unsigned short VmT[64 * 64];   // VmT[c][j] = V[j][c]
  __shared__ __align__(16) float Ss[64 * 64];
  const int tid  = threadIdx.x;
  const int lane = tid & 31;
  const int w    = tid >> 5;           // 4 waves, 16-row strips
  const int hi_  = blockIdx.x;         // head index (also the buggy "i")
  const int bb   = blockIdx.y;         // batch
  const size_t rowbase = (size_t)bb * 64 * DIMD + (size_t)hi_ * 64;

#pragma unroll
  for (int i = 0; i < 4; ++i) {        // Q,K head slices: 64 rows x 128B each
    int c = i * 128 + tid;
    int row = c >> 3, co = c & 7;
    async_b128(Q + rowbase + (size_t)row * DIMD + co * 8, &Qs[row * 64 + co * 8]);
    async_b128(K + rowbase + (size_t)row * DIMD + co * 8, &Ks[row * 64 + co * 8]);
  }
  {                                    // stage V transposed for the B-operand of P*V
    int j = tid >> 1, cb = (tid & 1) * 32;
    const unsigned short* src = V + rowbase + (size_t)j * DIMD + cb;
    for (int kk = 0; kk < 32; ++kk) VmT[(cb + kk) * 64 + j] = src[kk];
  }
  wait_async<0>();
  __syncthreads();

  // scores strip [w*16, w*16+16) x 64
  v8f sc[4] = {};
#pragma unroll
  for (int ks = 0; ks < 2; ++ks) {
    const int kb8  = ks * 32 + (lane >> 4) * 8;
    const int kb16 = ks * 32 + (lane >> 4) * 16;
    v16bf af = frag_a(&Qs[(w * 16 + (lane & 15)) * 64 + kb16]);
#pragma unroll
    for (int t = 0; t < 4; ++t) {
      v16bf bf = frag_b(&Ks[(t * 16 + (lane & 15)) * 64 + kb16]);
      sc[t] = __builtin_amdgcn_wmma_f32_16x16x32_bf16(false, af, false, bf, (short)0, sc[t], false, false);
    }
  }
#pragma unroll
  for (int t = 0; t < 4; ++t)
#pragma unroll
    for (int r = 0; r < 8; ++r) {
      int q = w * 16 + ((lane >> 4) << 3) + r;
      int c = t * 16 + (lane & 15);
      Ss[q * 64 + c] = sc[t][r];
    }
  __syncthreads();

  if (tid < 64) {                      // row softmax (no scaling, faithful)
    float m = -1e30f;
    for (int j = 0; j < 64; ++j) m = fmaxf(m, Ss[tid * 64 + j]);
    float s = 0.f;
    for (int j = 0; j < 64; ++j) { float e = __expf(Ss[tid * 64 + j] - m); Ss[tid * 64 + j] = e; s += e; }
    float inv = 1.f / s;
    for (int j = 0; j < 64; ++j) Ps[tid * 64 + j] = f2bf(Ss[tid * 64 + j] * inv);
  }
  __syncthreads();

  // out strip = P * V
  v8f oc[4] = {};
#pragma unroll
  for (int ks = 0; ks < 2; ++ks) {
    const int kb8  = ks * 32 + (lane >> 4) * 8;
    const int kb16 = ks * 32 + (lane >> 4) * 16;
    v16bf af = frag_a(&Ps[(w * 16 + (lane & 15)) * 64 + kb16]);
#pragma unroll
    for (int t = 0; t < 4; ++t) {
      v16bf bf = frag_b(&VmT[(t * 16 + (lane & 15)) * 64 + kb16]);
      oc[t] = __builtin_amdgcn_wmma_f32_16x16x32_bf16(false, af, false, bf, (short)0, oc[t], false, false);
    }
  }
  // scatter per cat(dim0)+transpose+view regrouping
#pragma unroll
  for (int t = 0; t < 4; ++t)
#pragma unroll
    for (int r = 0; r < 8; ++r) {
      int q = w * 16 + ((lane >> 4) << 3) + r;
      int c = t * 16 + (lane & 15);
      int b_o = (bb >> 1) + ((c >= 32) ? 32 : 0);
      int s_o = ((bb & 1) << 5) + (q >> 1);
      int d_o = ((q & 1) << 11) + hi_ * 32 + (c & 31);
      Y[(size_t)(b_o * 64 + s_o) * DIMD + d_o] = f2bf(oc[t][r]);
    }
}

extern "C" void kernel_launch(void* const* d_in, const int* in_sizes, int n_in,
                              void* d_out, int out_size, void* d_ws, size_t ws_size,
                              hipStream_t stream) {
  (void)in_sizes; (void)n_in; (void)out_size; (void)ws_size;
  const float* x  = (const float*)d_in[0];
  const float* Wq = (const float*)d_in[1];
  const float* bq = (const float*)d_in[2];
  const float* Wk = (const float*)d_in[3];
  const float* bk = (const float*)d_in[4];
  const float* Wv = (const float*)d_in[5];
  const float* bv = (const float*)d_in[6];
  const float* Wp = (const float*)d_in[7];
  const float* bp = (const float*)d_in[8];

  const size_t MAT = (size_t)DIMD * DIMD;   // 16,777,216 elements
  unsigned short* xb  = (unsigned short*)d_ws;
  unsigned short* wqb = xb  + MAT;
  unsigned short* wkb = wqb + MAT;
  unsigned short* wvb = wkb + MAT;
  unsigned short* wpb = wvb + MAT;
  unsigned short* Qb  = wpb + MAT;
  unsigned short* Kb  = Qb  + MAT;
  unsigned short* Vb  = Kb  + MAT;
  unsigned short* Yb  = Vb  + MAT;

  dim3 cblk(256), cgrd((unsigned)(MAT / (256 * 8)));
  hipLaunchKernelGGL(cvt_f32_bf16, cgrd, cblk, 0, stream, x,  xb,  (int)MAT);
  hipLaunchKernelGGL(cvt_f32_bf16, cgrd, cblk, 0, stream, Wq, wqb, (int)MAT);
  hipLaunchKernelGGL(cvt_f32_bf16, cgrd, cblk, 0, stream, Wk, wkb, (int)MAT);
  hipLaunchKernelGGL(cvt_f32_bf16, cgrd, cblk, 0, stream, Wv, wvb, (int)MAT);
  hipLaunchKernelGGL(cvt_f32_bf16, cgrd, cblk, 0, stream, Wp, wpb, (int)MAT);

  dim3 gb(256), gg(32, 32);                 // M/128 x N/128
  hipLaunchKernelGGL((gemm_xwt<true>),  gg, gb, 0, stream, xb, wqb, bq, (void*)Qb);
  hipLaunchKernelGGL((gemm_xwt<true>),  gg, gb, 0, stream, xb, wkb, bk, (void*)Kb);
  hipLaunchKernelGGL((gemm_xwt<true>),  gg, gb, 0, stream, xb, wvb, bv, (void*)Vb);

  dim3 ab(128), ag(NH, 64);                 // (head, batch)
  hipLaunchKernelGGL(attn_kernel, ag, ab, 0, stream, Qb, Kb, Vb, Yb);

  hipLaunchKernelGGL((gemm_xwt<false>), gg, gb, 0, stream, Yb, wpb, bp, d_out);
}
